// MultiHeadStrassenAttention_12541304504804
// MI455X (gfx1250) — matrix operand order repaired and checked by reading, not compiled, not run on hardware
//
#include <hip/hip_runtime.h>
#include <math.h>
#include <stdint.h>

// Problem constants (fixed by the reference)
#define HH 8
#define DD 64
#define NN 256
#define BB 2
#define CC 512
#define BH (BB*HH)     // 16
#define BN (BB*NN)     // 512
#define M5 (5*CC)      // 2560
#define SCALER 0.125f  // 1/sqrt(64)

#define Z8 {0.f,0.f,0.f,0.f,0.f,0.f,0.f,0.f}

typedef _Float16 f16;
typedef _Float16 v16h __attribute__((ext_vector_type(16)));
typedef _Float16 v8h  __attribute__((ext_vector_type(8)));
typedef float    v8f  __attribute__((ext_vector_type(8)));

static __device__ inline v16h cat16(v8h lo, v8h hi) {
  return __builtin_shufflevector(lo, hi, 0,1,2,3,4,5,6,7,8,9,10,11,12,13,14,15);
}

// A fragment (16x32 f16): lane L -> row r0+(L&15); K chunks kb..kb+7 and kb+16..kb+23, kb = kk + 8*(L>>4)
static __device__ inline v16h load_fragA(const f16* __restrict__ src, int ld, int r0, int kk, int lane) {
  const f16* p = src + (r0 + (lane & 15)) * ld + kk + ((lane >> 4) << 3);
  v8h lo = *(const v8h*)(p);
  v8h hi = *(const v8h*)(p + 16);
  return cat16(lo, hi);
}

// B fragment (32x16 f16) for C = A * B^T where Bt is row-major [N x K]:
// lane L -> column c0+(L&15); K chunks kb..kb+7 and kb+16..kb+23, kb = kk + 8*(L>>4)
static __device__ inline v16h load_fragB(const f16* __restrict__ src, int ld, int c0, int kk, int lane) {
  const f16* p = src + (c0 + (lane & 15)) * ld + kk + ((lane >> 4) << 3);
  v8h lo = *(const v8h*)(p);
  v8h hi = *(const v8h*)(p + 16);
  return cat16(lo, hi);
}

static __device__ inline v8f wmma16(v16h a, v16h b, v8f c) {
  return __builtin_amdgcn_wmma_f32_16x16x32_f16(false, a, false, b, (short)0, c, false, false);
}

// monotonic float<->int encoding for integer atomicMax
static __device__ inline int encf(float f) {
  int b = __float_as_int(f);
  return b ^ ((b >> 31) & 0x7FFFFFFF);
}
static __device__ inline float decf(int i) {
  int b = i ^ ((i >> 31) & 0x7FFFFFFF);
  return __int_as_float(b);
}

// CDNA5 async global->LDS copy (16B per lane), tracked by ASYNCcnt.
static __device__ inline void async_copy_b128(uint32_t lds_off, const void* gaddr) {
  uint64_t ga = (uint64_t)(uintptr_t)gaddr;
  asm volatile("global_load_async_to_lds_b128 %0, %1, off" :: "v"(lds_off), "v"(ga) : "memory");
}
static __device__ inline void wait_async0() {
  asm volatile("s_wait_asynccnt 0x0" ::: "memory");
}

// ---------------- Kernel 0: fp32 -> fp16 conversion of x and w ----------------
__global__ void k_convert(const float* __restrict__ x, const float* __restrict__ w,
                          f16* __restrict__ x16, f16* __restrict__ w16) {
  int i = blockIdx.x * blockDim.x + threadIdx.x;
  int stride = gridDim.x * blockDim.x;
  for (int t = i; t < BN * CC; t += stride) x16[t] = (f16)x[t];
  for (int t = i; t < M5 * CC; t += stride) w16[t] = (f16)w[t];
}

// ---------------- Kernel 1: projection GEMM y = x @ w^T, scattered into proj[5][bh][n][d]
// 1x2 register blocking: one A fragment feeds two WMMAs.
__global__ void __launch_bounds__(256) k_proj(const f16* __restrict__ x16,
                                              const f16* __restrict__ w16,
                                              f16* __restrict__ proj) {
  int lane = threadIdx.x & 31, wave = threadIdx.x >> 5;
  int bm = blockIdx.y;   // 0..7   (rows of 64 over BN=512)
  int bn2 = blockIdx.x;  // 0..39  (cols of 64 over M5=2560)
  int tr = wave & 3, tc2 = wave >> 2;           // 4 row-tiles x 2 column-pairs
  int r0 = bm * 64 + tr * 16;
  int c0 = bn2 * 64 + tc2 * 32;
  v8f acc0 = Z8, acc1 = Z8;
#pragma unroll
  for (int kk = 0; kk < CC; kk += 32) {
    v16h a  = load_fragA(x16, CC, r0, kk, lane);
    v16h b0 = load_fragB(w16, CC, c0,      kk, lane);
    v16h b1 = load_fragB(w16, CC, c0 + 16, kk, lane);
    acc0 = wmma16(a, b0, acc0);
    acc1 = wmma16(a, b1, acc1);
  }
#pragma unroll
  for (int q = 0; q < 2; q++) {
    v8f acc = q ? acc1 : acc0;
    int m = c0 + q * 16 + (lane & 15);
    int p = m >> 9, h = (m >> 6) & 7, d = m & 63;
    int rbase = r0 + ((lane >> 4) << 3);
#pragma unroll
    for (int v = 0; v < 8; v++) {
      int bn = rbase + v;
      int b_ = bn >> 8, n = bn & 255;
      proj[((size_t)((p * BB + b_) * HH + h) * NN + n) * DD + d] = (f16)acc[v];
    }
  }
}

// ---------------- Kernel 2a: raw scores (X=b@c^T, Yt=a@c^T, Z=a@b^T), scaled, + row maxes
__global__ void __launch_bounds__(256) k_scores(const f16* __restrict__ proj,
                                                float* __restrict__ scoreF32,
                                                float* __restrict__ rowmax) {
  int strip = blockIdx.x;  // 0..3  (64-row strip)
  int m     = blockIdx.y;  // 0..2  (which matrix)
  int bh    = blockIdx.z;  // 0..15
  int lane = threadIdx.x & 31, wave = threadIdx.x >> 5;

  const f16* pa = proj + (size_t)(0 * BH + bh) * NN * DD;
  const f16* pb = proj + (size_t)(1 * BH + bh) * NN * DD;
  const f16* pc = proj + (size_t)(2 * BH + bh) * NN * DD;
  const f16* A;  const f16* Bt;
  if (m == 0)      { A = pb; Bt = pc; }   // X[j][k]
  else if (m == 1) { A = pa; Bt = pc; }   // Yt[i][k]
  else             { A = pa; Bt = pb; }   // Z[i][j]

  __shared__ int rmaxI[64];
  if (threadIdx.x < 64) rmaxI[threadIdx.x] = (int)0x807FFFFF;  // enc(-inf)
  __syncthreads();

  int r0s = strip * 64;
  float* sb = scoreF32 + (size_t)(m * BH + bh) * NN * NN;

  for (int t = wave; t < 64; t += 8) {
    int r0 = r0s + (t >> 4) * 16;
    int c0 = (t & 15) * 16;
    v8f acc = Z8;
#pragma unroll
    for (int kk = 0; kk < DD; kk += 32) {
      v16h a = load_fragA(A, DD, r0, kk, lane);
      v16h b = load_fragB(Bt, DD, c0, kk, lane);
      acc = wmma16(a, b, acc);
    }
    int col = c0 + (lane & 15);
    int rb = r0 + ((lane >> 4) << 3);
#pragma unroll
    for (int v = 0; v < 8; v++) {
      float val = acc[v] * SCALER;
      sb[(size_t)(rb + v) * NN + col] = val;
      atomicMax(&rmaxI[rb + v - r0s], encf(val));
    }
  }
  __syncthreads();
  if (threadIdx.x < 64)
    rowmax[(m * BH + bh) * NN + r0s + threadIdx.x] = decf(rmaxI[threadIdx.x]);
}

// ---------------- Kernel 2b: exp(score - max) -> f16. X uses per-(b,h) global max, Yt/Z per-row.
__global__ void __launch_bounds__(256) k_softmax(const float* __restrict__ scoreF32,
                                                 const float* __restrict__ rowmax,
                                                 f16* __restrict__ scoreF16) {
  int strip = blockIdx.x, m = blockIdx.y, bh = blockIdx.z;
  int tid = threadIdx.x;
  __shared__ float red[256];
  float gmax = 0.f;
  if (m == 0) {
    red[tid] = rowmax[(0 * BH + bh) * NN + tid];
    __syncthreads();
    for (int s = 128; s > 0; s >>= 1) {
      if (tid < s) red[tid] = fmaxf(red[tid], red[tid + s]);
      __syncthreads();
    }
    gmax = red[0];
  }
  const float* sb = scoreF32 + (size_t)(m * BH + bh) * NN * NN + (size_t)strip * 64 * NN;
  f16* ob = scoreF16 + (size_t)(m * BH + bh) * NN * NN + (size_t)strip * 64 * NN;
  const float* rm = rowmax + (m * BH + bh) * NN + strip * 64;
  for (int i = 0; i < 64; i++) {
    float sub = (m == 0) ? gmax : rm[i];
    ob[(size_t)i * NN + tid] = (f16)__expf(sb[(size_t)i * NN + tid] - sub);
  }
}

// ---------------- Kernel 3 (heavy): per (bh, dslot in 0..64, quadrant):
//   V = (X .*cols vk_d) @ Yt^T restricted to a 128x128 quadrant (jhalf, ihalf),
//   NumPart[bh][dslot][jhalf][i] += sum_{j in half} Z[i][j] * vj_d[j] * V[j][i]
//   (dslot==64 -> vj=vk=1 -> Den). ~137KB LDS -> 2 workgroups per WGP.
__global__ void __launch_bounds__(256) k_heavy(const f16* __restrict__ proj,
                                               const f16* __restrict__ scoreF16,
                                               float* __restrict__ NumPart) {
  int dslot = blockIdx.x;        // 0..64
  int bh    = blockIdx.y;        // 0..15
  int quad  = blockIdx.z;        // 0..3
  int jhalf = quad >> 1, ihalf = quad & 1;
  int tid = threadIdx.x, lane = tid & 31, wave = tid >> 5;

  __shared__ __attribute__((aligned(16))) f16 Xs[128 * 264];  // 67584 B, 16B row pad
  __shared__ __attribute__((aligned(16))) f16 Ys[128 * 264];  // 67584 B
  __shared__ __attribute__((aligned(16))) f16 vkd[256];
  __shared__ __attribute__((aligned(16))) f16 vjd[256];
  __shared__ float numAcc[128];

  const f16* vjsrc = proj + (size_t)(3 * BH + bh) * NN * DD;
  const f16* vksrc = proj + (size_t)(4 * BH + bh) * NN * DD;
  vkd[tid] = (dslot == 64) ? (f16)1.0f : vksrc[tid * DD + dslot];
  vjd[tid] = (dslot == 64) ? (f16)1.0f : vjsrc[tid * DD + dslot];
  if (tid < 128) numAcc[tid] = 0.0f;

  const f16* Xg = scoreF16 + (size_t)(0 * BH + bh) * NN * NN + (size_t)jhalf * 128 * NN;
  const f16* Yg = scoreF16 + (size_t)(1 * BH + bh) * NN * NN + (size_t)ihalf * 128 * NN;
  const f16* Zg = scoreF16 + (size_t)(2 * BH + bh) * NN * NN;

  // Stage both 128x256 f16 half-matrices via ASYNC global->LDS copies (16B/lane).
  uint32_t xs_base = (uint32_t)(uintptr_t)(void*)Xs;  // LDS aperture: low 32 bits = LDS offset
  uint32_t ys_base = (uint32_t)(uintptr_t)(void*)Ys;
  for (int c = tid; c < 4096; c += 256) {   // 4096 chunks of 8 f16 per half-matrix
    int row = c >> 5, c16 = c & 31;
    uint32_t loff = (uint32_t)((row * 264 + c16 * 8) * 2);
    async_copy_b128(xs_base + loff, Xg + (size_t)row * NN + c16 * 8);
    async_copy_b128(ys_base + loff, Yg + (size_t)row * NN + c16 * 8);
  }
  wait_async0();
  __syncthreads();

  // Scale X columns by vk_d in place (vkd visible after the barrier above).
  for (int c = tid; c < 4096; c += 256) {
    int row = c >> 5, c16 = c & 31;
    v8h* px = (v8h*)(Xs + row * 264 + c16 * 8);
    v8h sc = *(const v8h*)(vkd + c16 * 8);
    *px = *px * sc;
  }
  __syncthreads();

  // 2x2 register tiling over the 128x128 quadrant: 16 groups of 32x32; 8 waves, 2 each.
  for (int g = wave; g < 16; g += 8) {
    int j0 = (g >> 2) << 5;   // local j (0..96)
    int i0 = (g & 3) << 5;    // local i (0..96)
    v8f acc00 = Z8, acc01 = Z8, acc10 = Z8, acc11 = Z8;
#pragma unroll
    for (int kk = 0; kk < 256; kk += 32) {
      v16h a0 = load_fragA(Xs, 264, j0,      kk, lane);
      v16h a1 = load_fragA(Xs, 264, j0 + 16, kk, lane);
      v16h b0 = load_fragB(Ys, 264, i0,      kk, lane);
      v16h b1 = load_fragB(Ys, 264, i0 + 16, kk, lane);
      // Shape the schedule: let next-step DS reads overlap the WMMA group.
      __builtin_amdgcn_sched_group_barrier(0x100, 8, 0);  // 8 DS ops
      acc00 = wmma16(a0, b0, acc00);
      acc01 = wmma16(a0, b1, acc01);
      acc10 = wmma16(a1, b0, acc10);
      acc11 = wmma16(a1, b1, acc11);
      __builtin_amdgcn_sched_group_barrier(0x8, 4, 0);    // 4 WMMA ops
    }
    // Epilogue: lane holds column i, rows jl..jl+7 of V; contract with Z[i][j]*vj[j].
#pragma unroll
    for (int q = 0; q < 4; q++) {
      int jt = j0 + ((q >> 1) << 4);            // local j tile base
      int it = i0 + ((q & 1) << 4);             // local i tile base
      v8f acc = (q == 0) ? acc00 : (q == 1) ? acc01 : (q == 2) ? acc10 : acc11;
      int il = it + (lane & 15);                // local i
      int jl = jt + ((lane >> 4) << 3);         // local j base (8 rows)
      int ig = ihalf * 128 + il;                // global i
      int jg = jhalf * 128 + jl;                // global j
      v8h zr = *(const v8h*)(Zg + (size_t)ig * NN + jg);
      v8h vj = *(const v8h*)(vjd + jg);
      float part = 0.f;
#pragma unroll
      for (int v = 0; v < 8; v++)
        part += (float)zr[v] * (float)vj[v] * acc[v];
      atomicAdd(&numAcc[il], part);
    }
  }
  __syncthreads();
  if (tid < 128)
    NumPart[(((size_t)bh * 65 + dslot) * 2 + jhalf) * NN + ihalf * 128 + tid] = numAcc[tid];
}

// ---------------- Kernel 4: out[b][n][h*64+d] = (Num0+Num1) / (Den0+Den1 + 1e-9)
__global__ void k_final(const float* __restrict__ NumPart, float* __restrict__ out) {
  int idx = blockIdx.x * 256 + threadIdx.x;     // over B*N*C = 262144
  int c = idx & (CC - 1);
  int n = (idx >> 9) & (NN - 1);
  int b = idx >> 17;
  int h = c >> 6, d = c & 63;
  int bh = b * HH + h;
  size_t nb = ((size_t)bh * 65 + d) * 2;
  size_t db = ((size_t)bh * 65 + 64) * 2;
  float num = NumPart[nb * NN + n] + NumPart[(nb + 1) * NN + n];
  float den = NumPart[db * NN + n] + NumPart[(db + 1) * NN + n] + 1e-9f;
  out[idx] = num / den;
}

extern "C" void kernel_launch(void* const* d_in, const int* in_sizes, int n_in,
                              void* d_out, int out_size, void* d_ws, size_t ws_size,
                              hipStream_t stream) {
  (void)in_sizes; (void)n_in; (void)out_size; (void)ws_size;
  const float* x = (const float*)d_in[0];
  const float* w = (const float*)d_in[1];
  float* out = (float*)d_out;

  char* p = (char*)d_ws;
  auto alloc = [&](size_t bytes) -> char* {
    char* r = p;
    p += (bytes + 255) & ~(size_t)255;
    return r;
  };
  f16*   x16      = (f16*)  alloc((size_t)BN * CC * sizeof(f16));
  f16*   w16      = (f16*)  alloc((size_t)M5 * CC * sizeof(f16));
  f16*   proj     = (f16*)  alloc((size_t)5 * BH * NN * DD * sizeof(f16));
  float* scoreF32 = (float*)alloc((size_t)3 * BH * NN * NN * sizeof(float));
  float* rowmax   = (float*)alloc((size_t)3 * BH * NN * sizeof(float));
  f16*   scoreF16 = (f16*)  alloc((size_t)3 * BH * NN * NN * sizeof(f16));
  float* NumPart  = (float*)alloc((size_t)BH * 65 * 2 * NN * sizeof(float));

  k_convert<<<dim3(512), dim3(256), 0, stream>>>(x, w, x16, w16);
  k_proj   <<<dim3(M5 / 64, BN / 64), dim3(256), 0, stream>>>(x16, w16, proj);
  k_scores <<<dim3(4, 3, BH), dim3(256), 0, stream>>>(proj, scoreF32, rowmax);
  k_softmax<<<dim3(4, 3, BH), dim3(256), 0, stream>>>(scoreF32, rowmax, scoreF16);
  k_heavy  <<<dim3(65, BH, 4), dim3(256), 0, stream>>>(proj, scoreF16, NumPart);
  k_final  <<<dim3(BN * CC / 256), dim3(256), 0, stream>>>(NumPart, out);
}
